// ArcFace_45981919871044
// MI455X (gfx1250) — matrix operand order repaired and checked by reading, not compiled, not run on hardware
//
#include <hip/hip_runtime.h>
#include <hip/hip_bf16.h>
#include <stdint.h>

typedef _Float16 half16 __attribute__((ext_vector_type(16)));
typedef _Float16 half4  __attribute__((ext_vector_type(4)));
typedef float    float8 __attribute__((ext_vector_type(8)));
typedef float    float4v __attribute__((ext_vector_type(4)));

#define C_TOTAL 100000
#define NROWS   1024
#define DIM     512
#define BC      64
#define NBLK    ((C_TOTAL + BC - 1) / BC)   // 1563
#define NTROWS  512                          // rows per GEMM workgroup

#define ARC_S     64.0f
#define ARC_COSM  0.8775825618903728f
#define ARC_SINM  0.479425538604203f
#define ARC_MINC  (-0.8775825618903728f)
#define ARC_M     0.5f

// ---------------------------------------------------------------------------
// Kernel 1: L2-normalize features; emit fp32 copy + f16 WMMA-A-fragment pack.
// A-frag layout (16-bit A 16x32, ISA 7.12.2): element (m, kq) ->
//   lane = m + 16*((kq>>3)&1),  e = 8*(kq>>4) + (kq&7)
// Packed as fnpack[mt][kc][lane][16] so each lane reads 32 contiguous bytes.
// ---------------------------------------------------------------------------
__global__ __launch_bounds__(64) void arcface_prep(
    const float* __restrict__ feats, float* __restrict__ fnf32,
    _Float16* __restrict__ fnpack) {
  __shared__ float red[64];
  const int n = blockIdx.x;   // row
  const int t = threadIdx.x;  // 0..63, handles 8 k-values
  const float* src = feats + (size_t)n * DIM + t * 8;
  float4v a = *(const float4v*)(src);
  float4v b = *(const float4v*)(src + 4);
  float sq = a.x*a.x + a.y*a.y + a.z*a.z + a.w*a.w
           + b.x*b.x + b.y*b.y + b.z*b.z + b.w*b.w;
  red[t] = sq;
  __syncthreads();
  for (int s = 32; s > 0; s >>= 1) {
    if (t < s) red[t] += red[t + s];
    __syncthreads();
  }
  const float scale = 1.0f / fmaxf(sqrtf(red[0]), 1e-12f);
  a *= scale; b *= scale;

  float* dst32 = fnf32 + (size_t)n * DIM + t * 8;
  *(float4v*)(dst32)     = a;
  *(float4v*)(dst32 + 4) = b;

  const int mt = n >> 4, m = n & 15;
  const int k0 = t * 8;
  const int kc = k0 >> 5;          // 32-wide K chunk
  const int kq = k0 & 31;          // 0,8,16,24
  const int lane = m + (((kq >> 3) & 1) << 4);
  const int e0 = (kq >> 4) << 3;
  _Float16* dst = fnpack + (((size_t)mt * 16 + kc) * 32 + lane) * 16 + e0;
  dst[0] = (_Float16)a.x; dst[1] = (_Float16)a.y;
  dst[2] = (_Float16)a.z; dst[3] = (_Float16)a.w;
  dst[4] = (_Float16)b.x; dst[5] = (_Float16)b.y;
  dst[6] = (_Float16)b.z; dst[7] = (_Float16)b.w;
}

// ---------------------------------------------------------------------------
// Kernel 2: WMMA GEMM block -> per-(block,row) softmax stats (max, sum-exp).
// grid = (NBLK, 2). 512 threads = 16 waves; wave = 64 rows x 32 cols
// = 4 Mtiles x 2 Ntiles = 8 v8f accumulators.
// B-frag layout (16-bit B 32x16, same k order as the A-frag): element (ncol, kq):
//   lane = ncol + 16*((kq>>3)&1), e = 8*(kq>>4) + (kq&7).
// ---------------------------------------------------------------------------
struct __align__(32) GemmLds {
  _Float16 bstage[4 * 32 * 16];   // 4 Ntiles * 32 lanes * 16 halves = 4 KB
  float wsq[BC];
  float smax[2][NTROWS];
  float ssum[2][NTROWS];
};

__global__ __launch_bounds__(512) void arcface_gemm(
    const float* __restrict__ w, const _Float16* __restrict__ fnpack,
    float* __restrict__ blkmax, float* __restrict__ blksum) {
  __shared__ GemmLds lds;

  const int t    = threadIdx.x;
  const int lane = t & 31;
  const int wv   = t >> 5;   // 0..15
  const int g    = wv & 1;   // column group (32 cols each)
  const int rb   = wv >> 1;  // row band (64 rows each)
  const int blk  = blockIdx.x;
  const int c0   = blk * BC;
  const int n0   = blockIdx.y * NTROWS;  // global row base
  const int mt0  = n0 >> 4;

  // cooperative B staging indices: 64 rows x 8 threads (float4 each)
  const int srow = t >> 3;                 // w row within block: 0..63
  const int skk  = (t & 7) << 2;           // k offset within chunk: 0..28
  const int c_ld = c0 + srow;
  const bool valid_ld = (c_ld < C_TOTAL);
  const int s_lane = (srow & 15) + (((skk >> 3) & 1) << 4);
  const int s_nt   = srow >> 4;
  _Float16* bst = &lds.bstage[((s_nt * 32 + s_lane) * 16) + (((skk >> 4) << 3) + (skk & 7))];
  const float* wrow = w + (size_t)c_ld * DIM;

  if (t < BC) lds.wsq[t] = 0.0f;

  float8 acc[4][2];
#pragma unroll
  for (int i = 0; i < 4; ++i) { acc[i][0] = (float8)0.0f; acc[i][1] = (float8)0.0f; }

  float sq = 0.0f;
  for (int kc = 0; kc < 16; ++kc) {
    __syncthreads();
    float4v v = valid_ld ? *(const float4v*)(wrow + kc * 32 + skk) : (float4v)0.0f;
    sq += v.x * v.x + v.y * v.y + v.z * v.z + v.w * v.w;
    half4 h = { (_Float16)v.x, (_Float16)v.y, (_Float16)v.z, (_Float16)v.w };
    *(half4*)bst = h;
    __syncthreads();

    half16 bf0 = *(const half16*)&lds.bstage[((g * 2 + 0) * 32 + lane) * 16];
    half16 bf1 = *(const half16*)&lds.bstage[((g * 2 + 1) * 32 + lane) * 16];
#pragma unroll
    for (int i = 0; i < 4; ++i) {
      const int mt = mt0 + rb * 4 + i;
      half16 af = *(const half16*)&fnpack[(((size_t)mt * 16 + kc) * 32 + lane) * 16];
      acc[i][0] = __builtin_amdgcn_wmma_f32_16x16x32_f16(
          false, af, false, bf0, (short)0, acc[i][0], false, false);
      acc[i][1] = __builtin_amdgcn_wmma_f32_16x16x32_f16(
          false, af, false, bf1, (short)0, acc[i][1], false, false);
    }
  }

  __syncthreads();
  if (valid_ld) atomicAdd(&lds.wsq[srow], sq);  // ds_add_f32
  __syncthreads();

  // finalize: logit = S * clamp(acc / ||w_c||, -1, 1)
  const int nb = lane & 15;
  const float wn0 = sqrtf(lds.wsq[g * 32 + nb]);
  const float wn1 = sqrtf(lds.wsq[g * 32 + 16 + nb]);
  const bool v0 = ((c0 + g * 32 + nb) < C_TOTAL) && (wn0 > 0.0f);
  const bool v1 = ((c0 + g * 32 + 16 + nb) < C_TOTAL) && (wn1 > 0.0f);
  const float r0 = v0 ? (1.0f / wn0) : 0.0f;
  const float r1 = v1 ? (1.0f / wn1) : 0.0f;

#pragma unroll
  for (int i = 0; i < 4; ++i) {
#pragma unroll
    for (int j = 0; j < 8; ++j) {
      float l0 = fminf(fmaxf(acc[i][0][j] * r0, -1.0f), 1.0f) * ARC_S;
      float l1 = fminf(fmaxf(acc[i][1][j] * r1, -1.0f), 1.0f) * ARC_S;
      acc[i][0][j] = v0 ? l0 : -1e30f;
      acc[i][1][j] = v1 ? l1 : -1e30f;
    }
  }

  // per-row (max, sum-exp) across this wave's 32 columns
#pragma unroll
  for (int i = 0; i < 4; ++i) {
#pragma unroll
    for (int j = 0; j < 8; ++j) {
      float l0 = acc[i][0][j], l1 = acc[i][1][j];
      float mx = fmaxf(l0, l1);
      for (int msk = 1; msk < 16; msk <<= 1)
        mx = fmaxf(mx, __shfl_xor(mx, msk, 32));
      float e = __expf(l0 - mx) + __expf(l1 - mx);
      for (int msk = 1; msk < 16; msk <<= 1)
        e += __shfl_xor(e, msk, 32);
      if ((lane & 15) == 0) {
        int row = rb * 64 + i * 16 + j + ((lane >> 4) << 3);
        lds.smax[g][row] = mx;
        lds.ssum[g][row] = e;
      }
    }
  }
  __syncthreads();

  // combine the two column groups, write block stats
  if (t < NTROWS) {
    float m0 = lds.smax[0][t], m1 = lds.smax[1][t];
    float s0 = lds.ssum[0][t], s1 = lds.ssum[1][t];
    float m = fmaxf(m0, m1);
    float s = s0 * __expf(m0 - m) + s1 * __expf(m1 - m);
    size_t idx = (size_t)blk * NROWS + n0 + t;
    blkmax[idx] = m;
    blksum[idx] = s;
  }
}

// ---------------------------------------------------------------------------
// Kernel 3: per-row log-sum-exp over blocks, exact fp32 margin patch for the
// label column, mean NLL.
// ---------------------------------------------------------------------------
__global__ __launch_bounds__(1024) void arcface_final(
    const float* __restrict__ blkmax, const float* __restrict__ blksum,
    const float* __restrict__ fnf32, const float* __restrict__ w,
    const int* __restrict__ labels, float* __restrict__ out) {
  __shared__ float red[NROWS];
  const int n = threadIdx.x;

  float M = -3e30f, s = 0.0f;
  for (int b = 0; b < NBLK; ++b) {
    float m  = blkmax[(size_t)b * NROWS + n];
    float sb = blksum[(size_t)b * NROWS + n];
    if (m > M) { s = s * __expf(M - m) + sb; M = m; }
    else       { s += sb * __expf(m - M); }
  }

  // exact fp32 cosine at the label column
  const int lab = labels[n];
  const float* wr = w + (size_t)lab * DIM;
  const float* fr = fnf32 + (size_t)n * DIM;
  float dot = 0.0f, sw = 0.0f;
  for (int k = 0; k < DIM; k += 4) {
    float4v a = *(const float4v*)(wr + k);
    float4v f = *(const float4v*)(fr + k);
    dot += a.x * f.x + a.y * f.y + a.z * f.z + a.w * f.w;
    sw  += a.x * a.x + a.y * a.y + a.z * a.z + a.w * a.w;
  }
  float cosv = fminf(fmaxf(dot / fmaxf(sqrtf(sw), 1e-12f), -1.0f), 1.0f);
  float sinv = sqrtf(1.0f - cosv * cosv + 1e-5f);
  float ctm  = cosv * ARC_COSM - sinv * ARC_SINM;
  float tl   = (cosv > ARC_MINC) ? ctm : (cosv - sinv * ARC_M);
  float co_l = ARC_S * cosv;
  float t_l  = ARC_S * tl;

  float s_adj = s - __expf(co_l - M) + __expf(t_l - M);
  float nll = -(t_l - M - __logf(s_adj));

  red[n] = nll;
  __syncthreads();
  for (int st = 512; st > 0; st >>= 1) {
    if (n < st) red[n] += red[n + st];
    __syncthreads();
  }
  if (n == 0) out[0] = red[0] * (1.0f / (float)NROWS);
}

// ---------------------------------------------------------------------------
extern "C" void kernel_launch(void* const* d_in, const int* in_sizes, int n_in,
                              void* d_out, int out_size, void* d_ws, size_t ws_size,
                              hipStream_t stream) {
  const float* feats  = (const float*)d_in[0];   // [8,128,512]
  const float* w      = (const float*)d_in[1];   // [100000,512]
  const int*   labels = (const int*)d_in[2];     // [1024]
  float* out = (float*)d_out;

  uintptr_t p = (uintptr_t)d_ws;
  auto align256 = [](uintptr_t x) { return (x + 255) & ~(uintptr_t)255; };
  p = align256(p);
  _Float16* fnpack = (_Float16*)p; p += (size_t)NROWS * DIM * sizeof(_Float16); p = align256(p);
  float* fnf32     = (float*)p;    p += (size_t)NROWS * DIM * sizeof(float);    p = align256(p);
  float* blkmax    = (float*)p;    p += (size_t)NBLK * NROWS * sizeof(float);   p = align256(p);
  float* blksum    = (float*)p;    p += (size_t)NBLK * NROWS * sizeof(float);

  arcface_prep<<<NROWS, 64, 0, stream>>>(feats, fnf32, fnpack);
  dim3 grid(NBLK, NROWS / NTROWS);
  arcface_gemm<<<grid, 512, 0, stream>>>(w, fnpack, blkmax, blksum);
  arcface_final<<<1, NROWS, 0, stream>>>(blkmax, blksum, fnf32, w, labels, out);
}
